// CodecAttention_81114752352309
// MI455X (gfx1250) — matrix operand order repaired and checked by reading, not compiled, not run on hardware
//
#include <hip/hip_runtime.h>
#include <hip/hip_bf16.h>

// Problem constants (match reference)
#define BB    2
#define TT    2048
#define DIMM  1024
#define NH    16
#define HD    64
#define WIN   512

typedef __attribute__((ext_vector_type(16))) __bf16 v16bf;
typedef __attribute__((ext_vector_type(8)))  float  v8f;
typedef __attribute__((ext_vector_type(4)))  float  f32x4;   // trivial 16B load vehicle

union Frag { v16bf v; f32x4 f4[2]; };

__device__ __forceinline__ int imin(int a, int b) { return a < b ? a : b; }
__device__ __forceinline__ int imax(int a, int b) { return a > b ? a : b; }

// ---------------------------------------------------------------------------
// elementwise f32 -> bf16 cast
// ---------------------------------------------------------------------------
__global__ void k_cast_bf16(const float* __restrict__ in, __bf16* __restrict__ out, int n) {
    int i = blockIdx.x * blockDim.x + threadIdx.x;
    if (i < n) out[i] = (__bf16)in[i];
}

// ---------------------------------------------------------------------------
// RMSNorm over last dim (1024), optional extra scale folded in, bf16 output
// ---------------------------------------------------------------------------
__global__ void k_rmsnorm_cast(const float* __restrict__ X, const float* __restrict__ gamma,
                               __bf16* __restrict__ Y, float extra) {
    __shared__ float red[256];
    const int row = blockIdx.x;
    const float* x = X + (size_t)row * DIMM;
    float s = 0.f;
    for (int c = threadIdx.x; c < DIMM; c += 256) { float v = x[c]; s += v * v; }
    red[threadIdx.x] = s;
    __syncthreads();
    for (int off = 128; off > 0; off >>= 1) {
        if (threadIdx.x < off) red[threadIdx.x] += red[threadIdx.x + off];
        __syncthreads();
    }
    const float inv = rsqrtf(red[0] * (1.0f / DIMM) + 1e-6f) * extra;
    __bf16* y = Y + (size_t)row * DIMM;
    for (int c = threadIdx.x; c < DIMM; c += 256) y[c] = (__bf16)(x[c] * inv * gamma[c]);
}

// ---------------------------------------------------------------------------
// V [B,T,H,D] f32 -> VT [B,H,D,T] bf16  (makes P·V B-fragments contiguous)
// ---------------------------------------------------------------------------
__global__ void k_build_vt(const float* __restrict__ V, __bf16* __restrict__ VTt, int n) {
    int i = blockIdx.x * blockDim.x + threadIdx.x;
    if (i >= n) return;
    int d = i & (HD - 1);
    int h = (i >> 6) & (NH - 1);
    int t = (i >> 10) & (TT - 1);
    int b = i >> 21;
    VTt[(((size_t)(b * NH + h) * HD + d) * TT) + t] = (__bf16)V[i];
}

// ---------------------------------------------------------------------------
// C[M,N] f32 = A[M,K] bf16 * W[N,K]^T bf16   (both row-major, K contiguous)
// block = 4 waves, block tile 64x64, wave tile 32x32 (2x2 wmma), K step 32
// ---------------------------------------------------------------------------
__global__ __launch_bounds__(128) void k_gemm_bf16_nt(
    const __bf16* __restrict__ A, const __bf16* __restrict__ W,
    float* __restrict__ C, int M, int N, int K) {
    const int lane = threadIdx.x & 31;
    const int wave = threadIdx.x >> 5;
    const int half = lane >> 4;
    const int l15  = lane & 15;
    const int m0 = blockIdx.x * 64 + (wave >> 1) * 32;
    const int n0 = blockIdx.y * 64 + (wave & 1) * 32;

    v8f c00 = {0,0,0,0,0,0,0,0};
    v8f c01 = {0,0,0,0,0,0,0,0};
    v8f c10 = {0,0,0,0,0,0,0,0};
    v8f c11 = {0,0,0,0,0,0,0,0};

    for (int k0 = 0; k0 < K; k0 += 32) {
        Frag a0, a1, b0, b1;
        // A fragment: lane row = m + l15; elems i <- k0 + half*16 + i (contiguous 32B)
        const __bf16* ap0 = A + (size_t)(m0 + l15) * K + k0 + half * 16;
        const __bf16* ap1 = A + (size_t)(m0 + 16 + l15) * K + k0 + half * 16;
        a0.f4[0] = *(const f32x4*)(ap0);
        a0.f4[1] = *(const f32x4*)(ap0 + 8);
        a1.f4[0] = *(const f32x4*)(ap1);
        a1.f4[1] = *(const f32x4*)(ap1 + 8);
        // B fragment: lane col = n + l15; elems i <- k0 + half*16 + i (contiguous 32B)
        const __bf16* bp0 = W + (size_t)(n0 + l15) * K + k0 + half * 16;
        const __bf16* bp1 = W + (size_t)(n0 + 16 + l15) * K + k0 + half * 16;
        b0.f4[0] = ((const f32x4*)bp0)[0];
        b0.f4[1] = ((const f32x4*)bp0)[1];
        b1.f4[0] = ((const f32x4*)bp1)[0];
        b1.f4[1] = ((const f32x4*)bp1)[1];
        // next-iteration prefetch into WGP-level caches (locality 3 => WGP scope;
        // data is L2-resident, so SYS/GL2-only prefetch would be useless).
        // Speculative; OOB prefetches are silently dropped.
        __builtin_prefetch((const void*)(ap0 + 32), 0, 3);
        __builtin_prefetch((const void*)(bp0 + 32), 0, 3);

        c00 = __builtin_amdgcn_wmma_f32_16x16x32_bf16(false, a0.v, false, b0.v, (short)0, c00, false, false);
        c01 = __builtin_amdgcn_wmma_f32_16x16x32_bf16(false, a0.v, false, b1.v, (short)0, c01, false, false);
        c10 = __builtin_amdgcn_wmma_f32_16x16x32_bf16(false, a1.v, false, b0.v, (short)0, c10, false, false);
        c11 = __builtin_amdgcn_wmma_f32_16x16x32_bf16(false, a1.v, false, b1.v, (short)0, c11, false, false);
    }
    // C layout: lane holds col n = l15, rows r + 8*half
    #pragma unroll
    for (int r = 0; r < 8; ++r) {
        const int mr0 = m0 + r + 8 * half;
        const int mr1 = mr0 + 16;
        C[(size_t)mr0 * N + n0 + l15]      = c00[r];
        C[(size_t)mr0 * N + n0 + 16 + l15] = c01[r];
        C[(size_t)mr1 * N + n0 + l15]      = c10[r];
        C[(size_t)mr1 * N + n0 + 16 + l15] = c11[r];
    }
}

// ---------------------------------------------------------------------------
// Sliding-window flash attention with ALiBi.
// grid = (T/64, H, B); block = 128 (4 waves); each wave owns 16 query rows.
// Scale 1/sqrt(D) is pre-folded into Qn.
// ---------------------------------------------------------------------------
__global__ __launch_bounds__(128) void k_attn(
    const __bf16* __restrict__ Qn, const __bf16* __restrict__ Kn,
    const __bf16* __restrict__ VTt, __bf16* __restrict__ Ob) {
    __shared__ __align__(16) __bf16 lds[4][16][40];   // per-wave P transpose tile

    const int lane = threadIdx.x & 31;
    const int wave = threadIdx.x >> 5;
    const int half = lane >> 4;
    const int l15  = lane & 15;
    const int h = blockIdx.y;
    const int b = blockIdx.z;
    const int qs = blockIdx.x * 64 + wave * 16;
    const float slope = exp2f(-0.5f * (float)(h + 1));

    // Q A-fragments for d=0..31 and d=32..63 (loaded once, reused all tiles)
    Frag qa0, qa1;
    {
        const __bf16* qp = Qn + (size_t)(b * TT + qs + l15) * DIMM + h * HD + half * 16;
        qa0.f4[0] = *(const f32x4*)(qp);
        qa0.f4[1] = *(const f32x4*)(qp + 8);
        qa1.f4[0] = *(const f32x4*)(qp + 32);
        qa1.f4[1] = *(const f32x4*)(qp + 40);
    }

    v8f o0 = {0,0,0,0,0,0,0,0};
    v8f o1 = {0,0,0,0,0,0,0,0};
    v8f o2 = {0,0,0,0,0,0,0,0};
    v8f o3 = {0,0,0,0,0,0,0,0};
    float mrow[8], lrow[8];
    #pragma unroll
    for (int r = 0; r < 8; ++r) { mrow[r] = -1e30f; lrow[r] = 0.f; }

    const __bf16* vbase = VTt + (size_t)(b * NH + h) * HD * TT;

    for (int kt = 0; kt < 17; ++kt) {
        const int jb = qs - WIN + kt * 32;   // jb is a multiple of 16

        // ---- S = Q K^T for keys [jb, jb+32) : two 16x16 tiles, K=64 split ----
        Frag kb00, kb01, kb10, kb11;
        {
            const int j0c = imin(imax(jb + l15, 0), TT - 1);        // clamped rows are
            const int j1c = imin(imax(jb + 16 + l15, 0), TT - 1);   // always masked below
            const __bf16* kp0 = Kn + (size_t)(b * TT + j0c) * DIMM + h * HD + half * 16;
            const __bf16* kp1 = Kn + (size_t)(b * TT + j1c) * DIMM + h * HD + half * 16;
            kb00.f4[0] = ((const f32x4*)kp0)[0];        kb00.f4[1] = ((const f32x4*)kp0)[1];
            kb01.f4[0] = ((const f32x4*)(kp0 + 32))[0]; kb01.f4[1] = ((const f32x4*)(kp0 + 32))[1];
            kb10.f4[0] = ((const f32x4*)kp1)[0];        kb10.f4[1] = ((const f32x4*)kp1)[1];
            kb11.f4[0] = ((const f32x4*)(kp1 + 32))[0]; kb11.f4[1] = ((const f32x4*)(kp1 + 32))[1];
            // prefetch next key tile's K rows (j + 32 => +32 rows) into WGP caches
            __builtin_prefetch((const void*)(kp0 + 32 * DIMM), 0, 3);
            __builtin_prefetch((const void*)(kp1 + 32 * DIMM), 0, 3);
        }
        v8f s0 = {0,0,0,0,0,0,0,0};
        v8f s1 = {0,0,0,0,0,0,0,0};
        s0 = __builtin_amdgcn_wmma_f32_16x16x32_bf16(false, qa0.v, false, kb00.v, (short)0, s0, false, false);
        s0 = __builtin_amdgcn_wmma_f32_16x16x32_bf16(false, qa1.v, false, kb01.v, (short)0, s0, false, false);
        s1 = __builtin_amdgcn_wmma_f32_16x16x32_bf16(false, qa0.v, false, kb10.v, (short)0, s1, false, false);
        s1 = __builtin_amdgcn_wmma_f32_16x16x32_bf16(false, qa1.v, false, kb11.v, (short)0, s1, false, false);

        // ---- ALiBi bias + causal/window mask (exact f32 math) ----
        #pragma unroll
        for (int r = 0; r < 8; ++r) {
            const int i  = qs + r + 8 * half;
            const int j0 = jb + l15;
            const int j1 = jb + 16 + l15;
            const float v0 = s0[r] + slope * (float)(j0 - i);
            const float v1 = s1[r] + slope * (float)(j1 - i);
            const bool bad0 = (j0 < 0) | (j0 >= TT) | (j0 > i) | (i - j0 > WIN);
            const bool bad1 = (j1 < 0) | (j1 >= TT) | (j1 > i) | (i - j1 > WIN);
            s0[r] = bad0 ? -1e30f : v0;
            s1[r] = bad1 ? -1e30f : v1;
        }

        // ---- online softmax; rows shared by 16 lanes of a half-wave ----
        float scl[8];
        #pragma unroll
        for (int r = 0; r < 8; ++r) {
            float cm = fmaxf(s0[r], s1[r]);
            cm = fmaxf(cm, __shfl_xor(cm, 1, 32));
            cm = fmaxf(cm, __shfl_xor(cm, 2, 32));
            cm = fmaxf(cm, __shfl_xor(cm, 4, 32));
            cm = fmaxf(cm, __shfl_xor(cm, 8, 32));
            const float mnew = fmaxf(mrow[r], cm);
            scl[r]  = __expf(mrow[r] - mnew);
            mrow[r] = mnew;
            const float p0 = (s0[r] < -1e29f) ? 0.f : __expf(s0[r] - mnew);
            const float p1 = (s1[r] < -1e29f) ? 0.f : __expf(s1[r] - mnew);
            s0[r] = p0; s1[r] = p1;
            float ps = p0 + p1;
            ps += __shfl_xor(ps, 1, 32);
            ps += __shfl_xor(ps, 2, 32);
            ps += __shfl_xor(ps, 4, 32);
            ps += __shfl_xor(ps, 8, 32);
            lrow[r] = lrow[r] * scl[r] + ps;
        }
        #pragma unroll
        for (int r = 0; r < 8; ++r) {
            o0[r] *= scl[r]; o1[r] *= scl[r]; o2[r] *= scl[r]; o3[r] *= scl[r];
        }

        // ---- P: C-layout -> A-layout via LDS (per-wave slice) ----
        #pragma unroll
        for (int r = 0; r < 8; ++r) {
            lds[wave][r + 8 * half][l15]      = (__bf16)s0[r];
            lds[wave][r + 8 * half][16 + l15] = (__bf16)s1[r];
        }
        __syncthreads();   // uniform trip count across all 4 waves
        Frag pa;
        {
            const __bf16* lp = &lds[wave][l15][half * 16];
            pa.f4[0] = *(const f32x4*)(lp);
            pa.f4[1] = *(const f32x4*)(lp + 8);
        }
        __syncthreads();

        // ---- O += P * V : 4 wmma over the 64-wide head dim ----
        // V B-frag: lane col = d chunk + l15; 16 contiguous keys along T.
        const int tstart = imin(imax(jb + half * 16, 0), TT - 16); // clamp => P==0 keys
        #pragma unroll
        for (int nt = 0; nt < 4; ++nt) {
            Frag vb;
            const __bf16* vp = vbase + (size_t)(nt * 16 + l15) * TT + tstart;
            vb.f4[0] = ((const f32x4*)vp)[0];
            vb.f4[1] = ((const f32x4*)vp)[1];
            __builtin_prefetch((const void*)(vp + 32), 0, 3);  // next key tile along T
            if (nt == 0) o0 = __builtin_amdgcn_wmma_f32_16x16x32_bf16(false, pa.v, false, vb.v, (short)0, o0, false, false);
            if (nt == 1) o1 = __builtin_amdgcn_wmma_f32_16x16x32_bf16(false, pa.v, false, vb.v, (short)0, o1, false, false);
            if (nt == 2) o2 = __builtin_amdgcn_wmma_f32_16x16x32_bf16(false, pa.v, false, vb.v, (short)0, o2, false, false);
            if (nt == 3) o3 = __builtin_amdgcn_wmma_f32_16x16x32_bf16(false, pa.v, false, vb.v, (short)0, o3, false, false);
        }
    }

    // ---- epilogue: normalize and write bf16 [B,T,H*D] ----
    #pragma unroll
    for (int r = 0; r < 8; ++r) {
        const float inv = 1.f / lrow[r];
        const int row = qs + r + 8 * half;
        const size_t base = (size_t)(b * TT + row) * DIMM + h * HD;
        Ob[base + 0  + l15] = (__bf16)(o0[r] * inv);
        Ob[base + 16 + l15] = (__bf16)(o1[r] * inv);
        Ob[base + 32 + l15] = (__bf16)(o2[r] * inv);
        Ob[base + 48 + l15] = (__bf16)(o3[r] * inv);
    }
}

// ---------------------------------------------------------------------------
extern "C" void kernel_launch(void* const* d_in, const int* in_sizes, int n_in,
                              void* d_out, int out_size, void* d_ws, size_t ws_size,
                              hipStream_t stream) {
    const float* x  = (const float*)d_in[0];
    const float* wq = (const float*)d_in[1];
    const float* wk = (const float*)d_in[2];
    const float* wv = (const float*)d_in[3];
    const float* wo = (const float*)d_in[4];
    const float* qg = (const float*)d_in[5];
    const float* kg = (const float*)d_in[6];

    const int M  = BB * TT;           // 4096
    const int NK = DIMM;              // 1024
    const size_t actN = (size_t)M * NK;
    const size_t wN   = (size_t)NK * NK;

    char* ws = (char*)d_ws;
    size_t off = 0;
    auto carve = [&](size_t bytes) { char* p = ws + off; off += (bytes + 255) & ~(size_t)255; return p; };
    __bf16* xb    = (__bf16*)carve(actN * 2);
    __bf16* wqb   = (__bf16*)carve(wN * 2);
    __bf16* wkb   = (__bf16*)carve(wN * 2);
    __bf16* wvb   = (__bf16*)carve(wN * 2);
    __bf16* wob   = (__bf16*)carve(wN * 2);
    float*  qf    = (float*) carve(actN * 4);
    float*  kf    = (float*) carve(actN * 4);
    float*  vf    = (float*) carve(actN * 4);
    __bf16* qn    = (__bf16*)carve(actN * 2);
    __bf16* kn    = (__bf16*)carve(actN * 2);
    __bf16* vt    = (__bf16*)carve(actN * 2);
    __bf16* attnb = (__bf16*)carve(actN * 2);

    const int thr = 256;
    k_cast_bf16<<<(int)((actN + thr - 1) / thr), thr, 0, stream>>>(x,  xb,  (int)actN);
    k_cast_bf16<<<(int)((wN + thr - 1) / thr),   thr, 0, stream>>>(wq, wqb, (int)wN);
    k_cast_bf16<<<(int)((wN + thr - 1) / thr),   thr, 0, stream>>>(wk, wkb, (int)wN);
    k_cast_bf16<<<(int)((wN + thr - 1) / thr),   thr, 0, stream>>>(wv, wvb, (int)wN);
    k_cast_bf16<<<(int)((wN + thr - 1) / thr),   thr, 0, stream>>>(wo, wob, (int)wN);

    dim3 gg(M / 64, NK / 64, 1);
    k_gemm_bf16_nt<<<gg, 128, 0, stream>>>(xb, wqb, qf, M, NK, NK);
    k_gemm_bf16_nt<<<gg, 128, 0, stream>>>(xb, wkb, kf, M, NK, NK);
    k_gemm_bf16_nt<<<gg, 128, 0, stream>>>(xb, wvb, vf, M, NK, NK);

    k_rmsnorm_cast<<<M, 256, 0, stream>>>(qf, qg, qn, 0.125f);  // fold 1/sqrt(64)
    k_rmsnorm_cast<<<M, 256, 0, stream>>>(kf, kg, kn, 1.0f);
    k_build_vt<<<(int)((actN + thr - 1) / thr), thr, 0, stream>>>(vf, vt, (int)actN);

    k_attn<<<dim3(TT / 64, NH, BB), 128, 0, stream>>>(qn, kn, vt, attnb);

    k_gemm_bf16_nt<<<gg, 128, 0, stream>>>(attnb, wob, (float*)d_out, M, NK, NK);
}
